// CRF_31817117729025
// MI455X (gfx1250) — matrix operand order repaired and checked by reading, not compiled, not run on hardware
//
#include <hip/hip_runtime.h>

typedef __attribute__((ext_vector_type(16))) _Float16 v16h;
typedef __attribute__((ext_vector_type(8)))  _Float16 v8h;
typedef __attribute__((ext_vector_type(8)))  float    v8f;
typedef __attribute__((ext_vector_type(4)))  float    v4f;
typedef __attribute__((ext_vector_type(4)))  unsigned int u32x4;
typedef __attribute__((ext_vector_type(8)))  int      i32x8;
typedef __attribute__((ext_vector_type(4)))  int      i32x4;

#define BB 512
#define TT 1024
#define NN 64
#define BT 16          // batches per block (WMMA M)
#define EPAD 68        // emit/alpha LDS row stride in dwords (64 + 4 pad)

#define LOG2E 1.4426950408889634f
#define LN2   0.6931471805599453f

union U16 { v16h v; v8h h[2]; };

// ---------------------------------------------------------------------------
// Kernel 1: per-batch length from boolean mask (prefix mask => len = popcount)
// and zero the output accumulator (runs every call => graph-replay safe).
// ---------------------------------------------------------------------------
__global__ void crf_len_zero(const unsigned char* __restrict__ mask,
                             int* __restrict__ len, float* __restrict__ out) {
    int b = blockIdx.x * blockDim.x + threadIdx.x;
    if (b == 0) out[0] = 0.0f;
    if (b < BB) {
        const unsigned long long* row =
            (const unsigned long long*)(mask + (size_t)b * TT);
        int s = 0;
        #pragma unroll 4
        for (int i = 0; i < TT / 8; ++i) s += __popcll(row[i]);
        len[b] = s;
    }
}

// ---------------------------------------------------------------------------
// Kernel 2: gold-path score, accumulated as -score/B into out.
// ---------------------------------------------------------------------------
__global__ void crf_score(const float* __restrict__ emit,
                          const int*   __restrict__ target,
                          const int*   __restrict__ len,
                          const float* __restrict__ trans,
                          const float* __restrict__ strans,
                          const float* __restrict__ etrans,
                          float* __restrict__ out) {
    int b = blockIdx.x * blockDim.x + threadIdx.x;
    if (b >= BB) return;
    int L = len[b];
    const int* tgt = target + (size_t)b * TT;
    float acc = strans[tgt[0]];
    int prev = 0;
    for (int t = 0; t < L; ++t) {
        int c = tgt[t];
        acc += emit[((size_t)b * TT + t) * NN + c];
        if (t > 0) acc += trans[prev * NN + c];
        prev = c;
    }
    acc += etrans[tgt[L - 1]];
    atomicAdd(out, -acc * (1.0f / (float)BB));
}

// ---------------------------------------------------------------------------
// TDM: DMA 4 rows x 64 floats of emit (row stride TT*NN floats) into LDS,
// padding 4 dwords after each 64-dword row => LDS row stride EPAD dwords.
// 6-arg (clang-23/therock) builtin; descriptor per CDNA5 ISA.
// ---------------------------------------------------------------------------
__device__ __forceinline__ void tdm_load_rows4(const float* gsrc, unsigned lds_off) {
    unsigned long long ga = (unsigned long long)(size_t)gsrc;
    u32x4 g0;
    g0[0] = 1u;                                        // count=1, load, no gather
    g0[1] = lds_off;                                   // lds_addr
    g0[2] = (unsigned)ga;                              // global_addr[31:0]
    g0[3] = (unsigned)((ga >> 32) & 0x01FFFFFFu) | (2u << 30);  // addr[56:32] | type=2
    i32x8 g1;
    g1[0] = (2 << 16)                                  // data_size = 4B
          | (1 << 20)                                  // pad_enable
          | (5 << 22)                                  // pad_interval: 64 dwords
          | (3 << 25);                                 // pad_amount: 4 dwords
    g1[1] = (64 << 16);                                // tensor_dim0 = 64
    g1[2] = (4 << 16);                                 // tensor_dim1 = 4
    g1[3] = (64 << 16);                                // tile_dim0 = 64
    g1[4] = 4;                                         // tile_dim1 = 4, tile_dim2 = 0
    g1[5] = TT * NN;                                   // tensor_dim0_stride
    g1[6] = 0;
    g1[7] = 0;
    i32x4 z4 = {0, 0, 0, 0};                           // groups 2/3 unused (2D)
    i32x8 z8 = {0, 0, 0, 0, 0, 0, 0, 0};               // extra group (unused)
    __builtin_amdgcn_tensor_load_to_lds(g0, g1, z4, z4, z8, 0);
}

// One forward step; MASKED is a compile-time 0/1 literal so the unmasked
// region is pure straight-line code (no exec-mask branches).
#define CRF_STEP(t, MASKED) do {                                               \
    const int tn = ((t) + 1 < TT) ? ((t) + 1) : (t);                           \
    tdm_load_rows4(gEmitWave + (size_t)tn * NN,                                \
                   ldsEmitBase + (unsigned)(((t) + 1) & 1) * ldsBufStride);    \
    /* A1: partial row max (16 rows x 8 segments over 128 threads) */          \
    v4f x0 = *(const v4f*)&alphaBuf[rA][segA * 8];                             \
    v4f x1 = *(const v4f*)&alphaBuf[rA][segA * 8 + 4];                         \
    float pm = fmaxf(fmaxf(fmaxf(x0.x, x0.y), fmaxf(x0.z, x0.w)),              \
                     fmaxf(fmaxf(x1.x, x1.y), fmaxf(x1.z, x1.w)));             \
    pmaxBuf[rA][segA] = pm;                                                    \
    __syncthreads();                                                           \
    /* A2+A3: full row max (redundant per segment), exp, pack f16 */           \
    v4f q0 = *(const v4f*)&pmaxBuf[rA][0];                                     \
    v4f q1 = *(const v4f*)&pmaxBuf[rA][4];                                     \
    float m = fmaxf(fmaxf(fmaxf(q0.x, q0.y), fmaxf(q0.z, q0.w)),               \
                    fmaxf(fmaxf(q1.x, q1.y), fmaxf(q1.z, q1.w)));              \
    if (segA == 0) mrow[rA] = m;                                               \
    v8h ph;                                                                    \
    ph[0] = (_Float16)__builtin_amdgcn_exp2f((x0.x - m) * LOG2E);              \
    ph[1] = (_Float16)__builtin_amdgcn_exp2f((x0.y - m) * LOG2E);              \
    ph[2] = (_Float16)__builtin_amdgcn_exp2f((x0.z - m) * LOG2E);              \
    ph[3] = (_Float16)__builtin_amdgcn_exp2f((x0.w - m) * LOG2E);              \
    ph[4] = (_Float16)__builtin_amdgcn_exp2f((x1.x - m) * LOG2E);              \
    ph[5] = (_Float16)__builtin_amdgcn_exp2f((x1.y - m) * LOG2E);              \
    ph[6] = (_Float16)__builtin_amdgcn_exp2f((x1.z - m) * LOG2E);              \
    ph[7] = (_Float16)__builtin_amdgcn_exp2f((x1.w - m) * LOG2E);              \
    *(v8h*)&pBuf[rA][segA * 8] = ph;                                           \
    __builtin_amdgcn_s_wait_tensorcnt(1);   /* buf[t&1] landed (per wave) */   \
    __syncthreads();                        /* pBuf + emit tile visible */     \
    /* GEMM tile via WMMA (all 128 threads, EXEC all-ones) */                  \
    U16 a0, a1;                                                                \
    a0.h[0] = *(const v8h*)&pBuf[M][kOff];                                     \
    a0.h[1] = *(const v8h*)&pBuf[M][kOff + 16];                                \
    a1.h[0] = *(const v8h*)&pBuf[M][32 + kOff];                                \
    a1.h[1] = *(const v8h*)&pBuf[M][32 + kOff + 16];                           \
    v8f acc = {0.f, 0.f, 0.f, 0.f, 0.f, 0.f, 0.f, 0.f};                        \
    acc = __builtin_amdgcn_wmma_f32_16x16x32_f16(false, a0.v, false, bop0.v,   \
                                                 (short)0, acc, false, false); \
    acc = __builtin_amdgcn_wmma_f32_16x16x32_f16(false, a1.v, false, bop1.v,   \
                                                 (short)0, acc, false, false); \
    /* Epilogue: alpha' = m + log(S) + emit */                                 \
    v4f m0 = *(const v4f*)&mrow[rBase];                                        \
    v4f m1 = *(const v4f*)&mrow[rBase + 4];                                    \
    float mv[8] = {m0.x, m0.y, m0.z, m0.w, m1.x, m1.y, m1.z, m1.w};            \
    const float* eb = &emitLds[(t) & 1][0][0];                                 \
    _Pragma("unroll")                                                          \
    for (int r = 0; r < 8; ++r) {                                              \
        const int Mr = rBase + r;                                              \
        float e   = eb[Mr * EPAD + nIdx];                                      \
        float val = mv[r] + __builtin_amdgcn_logf(acc[r]) * LN2 + e;           \
        if (MASKED) alphaReg[r] = ((t) < lenR[r]) ? val : alphaReg[r];         \
        else        alphaReg[r] = val;                                         \
        alphaBuf[Mr][nIdx] = alphaReg[r];                                      \
    }                                                                          \
    __syncthreads();                                                           \
} while (0)

// ---------------------------------------------------------------------------
// Kernel 3: forward (alpha) recursion via exp-factorized logsumexp:
//   alpha'_j = m + log( sum_i exp(alpha_i - m) * E_ij ) + emit_j ,  E=exp(trans)
// (16 x 64) x (64 x 64) per step via 2x v_wmma_f32_16x16x32_f16 per wave.
// One block = 16 batches, 4 waves; wave w owns output columns [16w,16w+16)
// and DMAs emit rows [4w,4w+4) for the next step via the TDM (double buffer).
// ---------------------------------------------------------------------------
__global__ __launch_bounds__(128)
void crf_forward(const float* __restrict__ emit,
                 const float* __restrict__ trans,
                 const float* __restrict__ strans,
                 const float* __restrict__ etrans,
                 const int*   __restrict__ len,
                 float* __restrict__ out) {
    __shared__ __align__(16) float     alphaBuf[BT][EPAD];
    __shared__ __align__(16) float     emitLds[2][BT][EPAD];  // TDM-padded tiles
    __shared__ __align__(16) _Float16  pBuf[BT][80];
    __shared__ __align__(16) _Float16  eT[NN][NN];            // eT[n][k] = exp(trans[k][n])
    __shared__ __align__(16) float     pmaxBuf[BT][8];
    __shared__ __align__(16) float     mrow[BT];
    __shared__ int   lenTile[BT];
    __shared__ int   sMinLen;

    const int tid  = threadIdx.x;
    const int lane = tid & 31;
    const int b0   = blockIdx.x * BT;
    // Wave-uniform scalars for TDM descriptor construction.
    const int wavS = __builtin_amdgcn_readfirstlane(tid >> 5);
    const unsigned ldsEmitBase =
        (unsigned)(size_t)(&emitLds[0][0][0]) + (unsigned)(wavS * 4 * EPAD * 4);
    const unsigned ldsBufStride = (unsigned)(BT * EPAD * 4);
    const float* gEmitWave = emit + (size_t)(b0 + 4 * wavS) * TT * NN;

    // Kick off the DMA of the t=1 emit tile immediately (buffer 1).
    tdm_load_rows4(gEmitWave + (size_t)1 * NN, ldsEmitBase + 1u * ldsBufStride);

    // Stage E^T (f16) into LDS: one-time, reused all T steps.
    for (int e = tid; e < NN * NN; e += 128) {
        int i = e >> 6, j = e & 63;                 // trans[i][j]
        eT[j][i] = (_Float16)__builtin_amdgcn_exp2f(trans[e] * LOG2E);
    }
    // alpha0 = strans + emit[:,0,:]
    for (int e = tid; e < BT * NN; e += 128) {
        int r = e >> 6, j = e & 63;
        alphaBuf[r][j] = strans[j] + emit[(size_t)(b0 + r) * TT * NN + j];
    }
    if (tid < BT) lenTile[tid] = len[b0 + tid];
    if (tid == 0) {
        int mn = TT;
        for (int i = 0; i < BT; ++i) mn = min(mn, len[b0 + i]);
        sMinLen = mn;
    }
    __syncthreads();

    // B operands resident in VGPRs for the whole T loop.
    // Dense f16 B (32x16): lanes 0-15 hold K=0..7,16..23, lanes 16-31 K=8..15,24..31.
    const int wave  = tid >> 5;
    const int nIdx  = wave * 16 + (lane & 15);
    const int kBase = (lane < 16) ? 0 : 8;
    U16 bop0, bop1;
    bop0.h[0] = *(const v8h*)&eT[nIdx][kBase];
    bop0.h[1] = *(const v8h*)&eT[nIdx][kBase + 16];
    bop1.h[0] = *(const v8h*)&eT[nIdx][32 + kBase];
    bop1.h[1] = *(const v8h*)&eT[nIdx][32 + kBase + 16];

    const int M     = lane & 15;                 // A-operand row
    const int kOff  = (lane < 16) ? 0 : 8;       // f16 A layout: lanes16-31 start at K=8/24
    const int rBase = (lane < 16) ? 0 : 8;       // f32 D layout: lanes16-31 hold M=8..15

    // Per-lane state in D-layout: alpha values and row lengths.
    float alphaReg[8];
    int   lenR[8];
    #pragma unroll
    for (int r = 0; r < 8; ++r) {
        alphaReg[r] = alphaBuf[rBase + r][nIdx];
        lenR[r]     = lenTile[rBase + r];
    }

    const int rA   = tid & 15;        // phase-A row
    const int segA = tid >> 4;        // phase-A 8-column segment

    // Unmasked region: t < min(len) (>= T/2), no freeze possible.
    int tSplit = __builtin_amdgcn_readfirstlane(sMinLen);
    tSplit = (tSplit < 1) ? 1 : ((tSplit > TT) ? TT : tSplit);

    int t = 1;
    for (; t < tSplit; ++t) CRF_STEP(t, 0);
    for (; t < TT; ++t)     CRF_STEP(t, 1);

    // ---- Final: logZ_b = logsumexp(alpha + etrans), accumulate logZ/B ----
    if (tid < BT) {
        const int r = tid;
        float m = -1e30f;
        #pragma unroll 8
        for (int j = 0; j < NN; ++j) m = fmaxf(m, alphaBuf[r][j] + etrans[j]);
        float s = 0.0f;
        #pragma unroll 8
        for (int j = 0; j < NN; ++j)
            s += __builtin_amdgcn_exp2f((alphaBuf[r][j] + etrans[j] - m) * LOG2E);
        float lz = m + __builtin_amdgcn_logf(s) * LN2;
        atomicAdd(out, lz * (1.0f / (float)BB));
    }
}

// ---------------------------------------------------------------------------
extern "C" void kernel_launch(void* const* d_in, const int* in_sizes, int n_in,
                              void* d_out, int out_size, void* d_ws, size_t ws_size,
                              hipStream_t stream) {
    const float*         emit   = (const float*)d_in[0];
    const int*           target = (const int*)d_in[1];
    const unsigned char* mask   = (const unsigned char*)d_in[2];
    const float*         trans  = (const float*)d_in[3];
    const float*         strans = (const float*)d_in[4];
    const float*         etrans = (const float*)d_in[5];
    float* out = (float*)d_out;
    int*   len = (int*)d_ws;          // 512 ints of scratch

    crf_len_zero<<<2, 256, 0, stream>>>(mask, len, out);
    crf_score  <<<2, 256, 0, stream>>>(emit, target, len, trans, strans, etrans, out);
    crf_forward<<<BB / BT, 128, 0, stream>>>(emit, trans, strans, etrans, len, out);
}
